// SmileTeacher_52802327937164
// MI455X (gfx1250) — matrix operand order repaired and checked by reading, not compiled, not run on hardware
//
#include <hip/hip_runtime.h>
#include <hip/hip_bf16.h>
#include <math.h>

// ---------------- problem constants ----------------
static constexpr int Bc  = 64;
static constexpr int Sc  = 512;
static constexpr int Dc  = 768;
static constexpr int Hc  = 12;
static constexpr int NLc = 12;
static constexpr int DHc = 64;
static constexpr int NFc = 32;
static constexpr int OUTc = 512;
static constexpr int Mc  = Bc * Sc;            // 32768 tokens

typedef __attribute__((ext_vector_type(16))) __bf16 v16bf;
typedef __attribute__((ext_vector_type(8)))  __bf16 v8bf;
typedef __attribute__((ext_vector_type(8)))  float  v8f;
typedef __attribute__((ext_vector_type(4)))  unsigned int v4u;
typedef __attribute__((ext_vector_type(4)))  int v4i;
typedef __attribute__((ext_vector_type(8)))  int v8i;

__device__ __forceinline__ unsigned short f2bf(float f) {
    unsigned int u = __float_as_uint(f);
    unsigned int r = u + 0x7FFFu + ((u >> 16) & 1u);   // round-to-nearest-even
    return (unsigned short)(r >> 16);
}

// ---------------- TDM: load a rows x 32 (bf16) tile global -> LDS ----------------
// D# per CDNA5 ISA ch.8: group0 = {count/lds_addr/global_addr/type}, group1 = dims.
// Tile: dim0 (k) = 32 elems, dim1 = rows, row stride = strideElems, data_size = 2B.
// LDS receives the tile contiguously (x fastest then y) == [rows][32] layout.
__device__ __forceinline__ void tdm_load_tile(const unsigned short* gsrc,
                                              unsigned lds_addr,
                                              int rows, int strideElems) {
    unsigned long long ga = (unsigned long long)(uintptr_t)gsrc;
    v4u g0;
    g0[0] = 1u;                                            // count=1, user mode
    g0[1] = lds_addr;                                      // LDS byte address
    g0[2] = (unsigned)ga;                                  // global_addr[31:0]
    g0[3] = (unsigned)((ga >> 32) & 0x01FFFFFFu) | (2u << 30);  // global_addr[56:32], type=2
    unsigned td0 = (unsigned)strideElems;                  // tensor_dim0 (>= tile_dim0, no OOB)
    v8i g1;
    g1[0] = 0x00010000;                                    // wg_mask=0, data_size=1 (2B)
    g1[1] = (int)((td0 & 0xFFFFu) << 16);                  // [79:48] tensor_dim0 lo16
    g1[2] = (int)(((td0 >> 16) & 0xFFFFu) | 0xFFFF0000u);  // tensor_dim0 hi16 | tensor_dim1 lo16 (huge)
    g1[3] = (int)((32u << 16) | 0x7FFFu);                  // tensor_dim1 hi16 | tile_dim0=32
    g1[4] = (int)((unsigned)rows & 0xFFFFu);               // tile_dim1=rows, tile_dim2=0
    g1[5] = (int)(unsigned)strideElems;                    // tensor_dim0_stride lo32
    g1[6] = 0;                                             // stride hi16 | dim1_stride lo16
    g1[7] = 0;
    v4i z4 = {0, 0, 0, 0};
#if defined(__clang_major__) && (__clang_major__ >= 23)
    v8i z8 = {0, 0, 0, 0, 0, 0, 0, 0};
    __builtin_amdgcn_tensor_load_to_lds(g0, g1, z4, z4, z8, 0);
#else
    __builtin_amdgcn_tensor_load_to_lds(g0, g1, z4, z4, 0);
#endif
}

// ---------------- weight convert + transpose: Wt[l][n][k] = bf16(W[l][k][n]) ----------------
// grid (Dc/32, Dc/32, NL), block 256
__global__ void cvt_tr_kernel(const float* __restrict__ src, unsigned short* __restrict__ dst) {
    __shared__ unsigned short t[32][33];
    const int l  = blockIdx.z;
    const int k0 = blockIdx.y * 32;
    const int n0 = blockIdx.x * 32;
    const float* S = src + (size_t)l * Dc * Dc;
    unsigned short* D = dst + (size_t)l * Dc * Dc;
    const int tx = threadIdx.x & 31;
    const int ty = threadIdx.x >> 5;      // 0..7
    #pragma unroll
    for (int j = 0; j < 4; ++j) {
        int k = ty + j * 8;
        t[k][tx] = f2bf(S[(size_t)(k0 + k) * Dc + n0 + tx]);   // coalesced read along n
    }
    __syncthreads();
    #pragma unroll
    for (int j = 0; j < 4; ++j) {
        int n = ty + j * 8;
        D[(size_t)(n0 + n) * Dc + k0 + tx] = t[tx][n];         // coalesced write along k
    }
}

// ---------------- embedding gather ----------------
__global__ void embed_kernel(const int* __restrict__ x, const float* __restrict__ emb,
                             float* __restrict__ h32, unsigned short* __restrict__ h16, int n) {
    int i = blockIdx.x * 256 + threadIdx.x;
    if (i >= n) return;
    int tok = i / Dc, d = i % Dc;
    float v = emb[(size_t)x[tok] * Dc + d];
    h32[i] = v;
    h16[i] = f2bf(v);
}

// ---------------- WMMA bf16 GEMM: C[M,N] = A[M,K] @ Wt^T + bias (Wt is N x K) ----------------
// block tile 128x256, 8 waves (2 m x 4 n), wave tile 64x64 = 4x4 wmma, K step 32.
// A/B tiles staged by the Tensor Data Mover (double-buffered), not by VALU threads.
template <int ACT>   // 0 = none, 1 = exact gelu
__global__ __launch_bounds__(256)
void gemm_bf16_kernel(const unsigned short* __restrict__ A,    // M x K row-major bf16
                      const unsigned short* __restrict__ Wt,   // N x K row-major bf16 (transposed weight)
                      const float* __restrict__ bias,
                      float* __restrict__ outF,
                      unsigned short* __restrict__ outH,
                      int M, int N, int K) {
    __shared__ unsigned short As[2][128][32];   // m x k, double buffered (16 KB)
    __shared__ unsigned short Bs[2][256][32];   // n x k, double buffered (32 KB)

    const int tid  = threadIdx.x;
    const int lane = tid & 31;
    const int wid  = tid >> 5;
    const int wm   = (wid & 1) * 64;         // 2 wave rows
    const int wn   = (wid >> 1) * 64;        // 4 wave cols
    const int lrow = lane & 15;
    const int lhi  = lane >> 4;
    const int mb   = blockIdx.y * 128;
    const int nb   = blockIdx.x * 256;

    v8f acc[4][4];
    #pragma unroll
    for (int i = 0; i < 4; ++i)
        #pragma unroll
        for (int j = 0; j < 4; ++j)
            acc[i][j] = 0.0f;                // scalar splat -> zero vector

    const unsigned short* aBase = A  + (size_t)mb * K;
    const unsigned short* bBase = Wt + (size_t)nb * K;
    const int nIter = K / 32;

    // prime buffer 0 via TDM (one wave issues; TDM ignores EXEC, per-wave issue)
    if (wid == 0) {
        tdm_load_tile(aBase, (unsigned)(uintptr_t)&As[0][0][0], 128, K);
        tdm_load_tile(bBase, (unsigned)(uintptr_t)&Bs[0][0][0], 256, K);
    }

    for (int it = 0; it < nIter; ++it) {
        const int cur = it & 1;
        if (wid == 0) {
            if (it + 1 < nIter) {
                // issue next tiles into the other buffer, then retire current pair
                tdm_load_tile(aBase + (it + 1) * 32, (unsigned)(uintptr_t)&As[cur ^ 1][0][0], 128, K);
                tdm_load_tile(bBase + (it + 1) * 32, (unsigned)(uintptr_t)&Bs[cur ^ 1][0][0], 256, K);
                __builtin_amdgcn_s_wait_tensorcnt((short)2);
            } else {
                __builtin_amdgcn_s_wait_tensorcnt((short)0);
            }
        }
        __syncthreads();

        // A fragments: k(e) = (e/8)*16 + lhi*8 + e%8  (ISA 16-bit A layout)
        v16bf af[4];
        #pragma unroll
        for (int tm = 0; tm < 4; ++tm) {
            int r = wm + tm * 16 + lrow;
            v8bf lo = *(const v8bf*)&As[cur][r][lhi * 8];
            v8bf hi = *(const v8bf*)&As[cur][r][16 + lhi * 8];
            #pragma unroll
            for (int e = 0; e < 8; ++e) { af[tm][e] = lo[e]; af[tm][e + 8] = hi[e]; }
        }
        // B fragments: n = lane%16, k(e) = (e/8)*16 + lhi*8 + e%8
        v16bf bfr[4];
        #pragma unroll
        for (int tn = 0; tn < 4; ++tn) {
            int c = wn + tn * 16 + lrow;
            v8bf lo = *(const v8bf*)&Bs[cur][c][lhi * 8];
            v8bf hi = *(const v8bf*)&Bs[cur][c][16 + lhi * 8];
            #pragma unroll
            for (int e = 0; e < 8; ++e) { bfr[tn][e] = lo[e]; bfr[tn][e + 8] = hi[e]; }
        }
        // 16 WMMAs per K-step per wave
        #pragma unroll
        for (int tm = 0; tm < 4; ++tm)
            #pragma unroll
            for (int tn = 0; tn < 4; ++tn)
                acc[tm][tn] = __builtin_amdgcn_wmma_f32_16x16x32_bf16(
                    false, af[tm], false, bfr[tn], (short)0, acc[tm][tn], false, false);
        __syncthreads();   // protects buffer reuse two iterations ahead
    }

    // epilogue: C layout -> n = lane%16, m = vgpr r + 8*(lane/16)
    #pragma unroll
    for (int tm = 0; tm < 4; ++tm)
        #pragma unroll
        for (int tn = 0; tn < 4; ++tn) {
            int n = nb + wn + tn * 16 + lrow;
            float bv = bias[n];
            #pragma unroll
            for (int r = 0; r < 8; ++r) {
                int m = mb + wm + tm * 16 + lhi * 8 + r;
                float v = acc[tm][tn][r] + bv;
                if (ACT == 1) v = 0.5f * v * (1.0f + erff(v * 0.70710678118654752f));
                size_t o = (size_t)m * N + n;
                if (outF) outF[o] = v;
                if (outH) outH[o] = f2bf(v);
            }
        }
}

// ---------------- RoPE + feature map: p = relu(rope(q) @ omega) [* mask] ----------------
__global__ void rope_feat_kernel(const float* __restrict__ qkin,
                                 const float* __restrict__ omega,
                                 const float* __restrict__ mask,
                                 float* __restrict__ outP, int use_mask) {
    int idx = blockIdx.x;                 // (b*S+s)*H + h
    int bs  = idx / Hc;
    int s   = bs & (Sc - 1);
    int d   = threadIdx.x;                // 0..63
    __shared__ float sq[DHc];
    size_t base = (size_t)idx * DHc;
    float qv = qkin[base + d];
    float pr = qkin[base + ((d < 32) ? d + 32 : d - 32)];
    float invf = powf(10000.0f, -(float)(2 * (d & 31)) / 64.0f);
    float ang = (float)s * invf;
    float sn, cs;
    sincosf(ang, &sn, &cs);
    sq[d] = qv * cs + ((d < 32) ? -pr : pr) * sn;
    __syncthreads();
    if (d < NFc) {
        float acc = 0.0f;
        #pragma unroll 8
        for (int k = 0; k < DHc; ++k) acc += sq[k] * omega[k * NFc + d];
        acc = fmaxf(acc, 0.0f);
        if (use_mask) acc *= mask[bs];
        outP[(size_t)idx * NFc + d] = acc;
    }
}

// ---------------- ksum[b,h,f] = sum_s pk ----------------
__global__ void ksum_kernel(const float* __restrict__ pk, float* __restrict__ ksum) {
    int bh = blockIdx.x, f = threadIdx.x;
    int b = bh / Hc, h = bh % Hc;
    float acc = 0.0f;
    for (int s = 0; s < Sc; ++s)
        acc += pk[(((size_t)(b * Sc + s)) * Hc + h) * NFc + f];
    ksum[(size_t)bh * NFc + f] = acc;
}

// ---------------- kv[b,h,f,d] = sum_s pk * v ----------------
__global__ __launch_bounds__(256)
void kv_kernel(const float* __restrict__ pk, const float* __restrict__ v,
               float* __restrict__ kv) {
    int bh = blockIdx.x;
    int b = bh / Hc, h = bh % Hc;
    int t = threadIdx.x;
    int d  = t & 63;
    int fg = t >> 6;                       // 0..3 -> 8 f each
    float acc[8];
    #pragma unroll
    for (int j = 0; j < 8; ++j) acc[j] = 0.0f;
    __shared__ float sPk[NFc];
    __shared__ float sV[DHc];
    for (int s = 0; s < Sc; ++s) {
        size_t tb = (size_t)(b * Sc + s) * Hc + h;
        if (t < 32) sPk[t] = pk[tb * NFc + t];
        else if (t < 96) sV[t - 32] = v[tb * DHc + (t - 32)];
        __syncthreads();
        float vv = sV[d];
        #pragma unroll
        for (int j = 0; j < 8; ++j) acc[j] += sPk[fg * 8 + j] * vv;
        __syncthreads();
    }
    #pragma unroll
    for (int j = 0; j < 8; ++j)
        kv[((size_t)bh * NFc + fg * 8 + j) * DHc + d] = acc[j];
}

// ---------------- att = (pq @ kv) * z, emitted as bf16 (feeds Wo GEMM) ----------------
__global__ void att_kernel(const float* __restrict__ pq, const float* __restrict__ kv,
                           const float* __restrict__ ksum, unsigned short* __restrict__ att16) {
    int idx = blockIdx.x;                 // (b*S+s)*H + h
    int h = idx % Hc;
    int bs = idx / Hc;
    int b = bs >> 9;
    int d = threadIdx.x;                  // 0..63
    __shared__ float spq[NFc];
    if (d < NFc) spq[d] = pq[(size_t)idx * NFc + d];
    __syncthreads();
    int bh = b * Hc + h;
    float zden = 1e-6f, acc = 0.0f;
    #pragma unroll 8
    for (int f = 0; f < NFc; ++f) {
        float pf = spq[f];
        zden += pf * ksum[(size_t)bh * NFc + f];
        acc  += pf * kv[((size_t)bh * NFc + f) * DHc + d];
    }
    att16[(size_t)idx * DHc + d] = f2bf(acc / zden);
}

// ---------------- residual add + LayerNorm, writes f32 (in place) + bf16 ----------------
__global__ __launch_bounds__(256)
void ln_kernel(float* __restrict__ h, const float* __restrict__ add,
               const float* __restrict__ sc, const float* __restrict__ bi,
               unsigned short* __restrict__ h16) {
    __shared__ float red[256];
    const int t = blockIdx.x;
    const int tid = threadIdx.x;
    size_t base = (size_t)t * Dc;
    float v0 = h[base + tid]       + add[base + tid];
    float v1 = h[base + tid + 256] + add[base + tid + 256];
    float v2 = h[base + tid + 512] + add[base + tid + 512];
    red[tid] = v0 + v1 + v2;
    __syncthreads();
    for (int s = 128; s > 0; s >>= 1) {
        if (tid < s) red[tid] += red[tid + s];
        __syncthreads();
    }
    float mean = red[0] * (1.0f / 768.0f);
    __syncthreads();
    float d0 = v0 - mean, d1 = v1 - mean, d2 = v2 - mean;
    red[tid] = d0 * d0 + d1 * d1 + d2 * d2;
    __syncthreads();
    for (int s = 128; s > 0; s >>= 1) {
        if (tid < s) red[tid] += red[tid + s];
        __syncthreads();
    }
    float rstd = rsqrtf(red[0] * (1.0f / 768.0f) + 1e-5f);
    float o0 = d0 * rstd * sc[tid]       + bi[tid];
    float o1 = d1 * rstd * sc[tid + 256] + bi[tid + 256];
    float o2 = d2 * rstd * sc[tid + 512] + bi[tid + 512];
    h[base + tid] = o0;        h16[base + tid] = f2bf(o0);
    h[base + tid + 256] = o1;  h16[base + tid + 256] = f2bf(o1);
    h[base + tid + 512] = o2;  h16[base + tid + 512] = f2bf(o2);
}

// ---------------- masked mean pooling ----------------
__global__ void pool_kernel(const float* __restrict__ h, const float* __restrict__ mask,
                            float* __restrict__ pooled) {
    int b = blockIdx.x;
    int d = threadIdx.x;                  // 0..767
    __shared__ float sm[Sc];
    if (d < Sc) sm[d] = mask[b * Sc + d];
    __syncthreads();
    float acc = 0.0f, den = 0.0f;
    for (int s = 0; s < Sc; ++s) {
        float mv = sm[s];
        acc += h[((size_t)b * Sc + s) * Dc + d] * mv;
        den += mv;
    }
    den = fmaxf(den, 1e-9f);
    pooled[(size_t)b * Dc + d] = acc / den;
}

// ---------------- final linear + relu ----------------
__global__ void final_kernel(const float* __restrict__ pooled, const float* __restrict__ Wl,
                             const float* __restrict__ bl, float* __restrict__ out) {
    int b = blockIdx.x;
    int o = threadIdx.x;                  // 0..511
    float acc = bl[o];
    #pragma unroll 4
    for (int d = 0; d < Dc; ++d)
        acc += pooled[(size_t)b * Dc + d] * Wl[(size_t)d * OUTc + o];
    out[(size_t)b * OUTc + o] = fmaxf(acc, 0.0f);
}

// ---------------- host orchestration ----------------
extern "C" void kernel_launch(void* const* d_in, const int* in_sizes, int n_in,
                              void* d_out, int out_size, void* d_ws, size_t ws_size,
                              hipStream_t stream) {
    (void)in_sizes; (void)n_in; (void)out_size; (void)ws_size;
    const int*   x      = (const int*)  d_in[0];
    const float* mask   = (const float*)d_in[1];
    const float* tokemb = (const float*)d_in[2];
    const float* Wq = (const float*)d_in[3];
    const float* bq = (const float*)d_in[4];
    const float* Wk = (const float*)d_in[5];
    const float* bk = (const float*)d_in[6];
    const float* Wv = (const float*)d_in[7];
    const float* bv = (const float*)d_in[8];
    const float* Wo = (const float*)d_in[9];
    const float* bo = (const float*)d_in[10];
    const float* omega = (const float*)d_in[11];
    const float* ln1s = (const float*)d_in[12];
    const float* ln1b = (const float*)d_in[13];
    const float* W1 = (const float*)d_in[14];
    const float* b1 = (const float*)d_in[15];
    const float* W2 = (const float*)d_in[16];
    const float* b2 = (const float*)d_in[17];
    const float* ln2s = (const float*)d_in[18];
    const float* ln2b = (const float*)d_in[19];
    const float* Wl = (const float*)d_in[20];
    const float* bl = (const float*)d_in[21];
    float* out = (float*)d_out;

    char* ws = (char*)d_ws;
    size_t off = 0;
    auto take = [&](size_t bytes) -> char* {
        char* p = ws + off;
        off += (bytes + 255) & ~(size_t)255;
        return p;
    };
    const size_t MD = (size_t)Mc * Dc;
    const size_t WSZ = (size_t)NLc * Dc * Dc;

    unsigned short* wq16 = (unsigned short*)take(WSZ * 2);   // all stored transposed: [l][n][k]
    unsigned short* wk16 = (unsigned short*)take(WSZ * 2);
    unsigned short* wv16 = (unsigned short*)take(WSZ * 2);
    unsigned short* wo16 = (unsigned short*)take(WSZ * 2);
    unsigned short* w116 = (unsigned short*)take(WSZ * 2);
    unsigned short* w216 = (unsigned short*)take(WSZ * 2);
    float*          h32  = (float*)take(MD * 4);
    unsigned short* h16  = (unsigned short*)take(MD * 2);
    float*          tmp32= (float*)take(MD * 4);
    float*          q32  = (float*)take(MD * 4);
    float*          k32  = (float*)take(MD * 4);
    float*          v32  = (float*)take(MD * 4);
    float*          pq   = (float*)take((size_t)Mc * Hc * NFc * 4);
    float*          pk   = (float*)take((size_t)Mc * Hc * NFc * 4);
    float*          kvb  = (float*)take((size_t)Bc * Hc * NFc * DHc * 4);
    float*          ksumb= (float*)take((size_t)Bc * Hc * NFc * 4);
    unsigned short* s16  = (unsigned short*)take(MD * 2);   // att bf16 / gelu bf16 (aliased)
    float*          pooled=(float*)take((size_t)Bc * Dc * 4);

    // weights -> bf16, transposed to [n][k] (once per call; deterministic)
    {
        dim3 tg(Dc / 32, Dc / 32, NLc);
        cvt_tr_kernel<<<tg, 256, 0, stream>>>(Wq, wq16);
        cvt_tr_kernel<<<tg, 256, 0, stream>>>(Wk, wk16);
        cvt_tr_kernel<<<tg, 256, 0, stream>>>(Wv, wv16);
        cvt_tr_kernel<<<tg, 256, 0, stream>>>(Wo, wo16);
        cvt_tr_kernel<<<tg, 256, 0, stream>>>(W1, w116);
        cvt_tr_kernel<<<tg, 256, 0, stream>>>(W2, w216);
    }
    // embedding
    {
        int n = (int)MD;
        embed_kernel<<<(n + 255) / 256, 256, 0, stream>>>(x, tokemb, h32, h16, n);
    }

    dim3 ggrid(Dc / 256, Mc / 128);   // 3 x 256 blocks
    for (int l = 0; l < NLc; ++l) {
        const size_t wOff = (size_t)l * Dc * Dc;
        const size_t bOff = (size_t)l * Dc;
        // QKV projections (bf16 WMMA, f32 out)
        gemm_bf16_kernel<0><<<ggrid, 256, 0, stream>>>(h16, wq16 + wOff, bq + bOff, q32, nullptr, Mc, Dc, Dc);
        gemm_bf16_kernel<0><<<ggrid, 256, 0, stream>>>(h16, wk16 + wOff, bk + bOff, k32, nullptr, Mc, Dc, Dc);
        gemm_bf16_kernel<0><<<ggrid, 256, 0, stream>>>(h16, wv16 + wOff, bv + bOff, v32, nullptr, Mc, Dc, Dc);
        // RoPE + feature map
        const float* om = omega + (size_t)l * DHc * NFc;
        rope_feat_kernel<<<Mc * Hc, 64, 0, stream>>>(q32, om, mask, pq, 0);
        rope_feat_kernel<<<Mc * Hc, 64, 0, stream>>>(k32, om, mask, pk, 1);
        // linear attention
        ksum_kernel<<<Bc * Hc, 32, 0, stream>>>(pk, ksumb);
        kv_kernel<<<Bc * Hc, 256, 0, stream>>>(pk, v32, kvb);
        att_kernel<<<Mc * Hc, 64, 0, stream>>>(pq, kvb, ksumb, s16);
        // output projection + residual LN1
        gemm_bf16_kernel<0><<<ggrid, 256, 0, stream>>>(s16, wo16 + wOff, bo + bOff, tmp32, nullptr, Mc, Dc, Dc);
        ln_kernel<<<Mc, 256, 0, stream>>>(h32, tmp32, ln1s + bOff, ln1b + bOff, h16);
        // FFN: gelu(h@W1+b1)@W2+b2, residual LN2
        gemm_bf16_kernel<1><<<ggrid, 256, 0, stream>>>(h16, w116 + wOff, b1 + bOff, nullptr, s16, Mc, Dc, Dc);
        gemm_bf16_kernel<0><<<ggrid, 256, 0, stream>>>(s16, w216 + wOff, b2 + bOff, tmp32, nullptr, Mc, Dc, Dc);
        ln_kernel<<<Mc, 256, 0, stream>>>(h32, tmp32, ln2s + bOff, ln2b + bOff, h16);
    }

    pool_kernel<<<Bc, Dc, 0, stream>>>(h32, mask, pooled);
    final_kernel<<<Bc, OUTc, 0, stream>>>(pooled, Wl, bl, out);
}
